// LINKX_24481313587823
// MI455X (gfx1250) — matrix operand order repaired and checked by reading, not compiled, not run on hardware
//
#include <hip/hip_runtime.h>

typedef __attribute__((ext_vector_type(16))) __bf16 bf16x16;
typedef __attribute__((ext_vector_type(8)))  float  v8f;

// ---------------------------------------------------------------- utilities
__global__ void k_zero(float* __restrict__ p, long n) {
    long i = (long)blockIdx.x * blockDim.x + threadIdx.x;
    long s = (long)gridDim.x * blockDim.x;
    for (; i < n; i += s) p[i] = 0.0f;
}

// ---------------------------------------------------------------- graph part
__global__ void k_degree(const int* __restrict__ col, float* __restrict__ deg, int E) {
    int e = blockIdx.x * blockDim.x + threadIdx.x;
    if (e < E) atomicAdd(&deg[col[e]], 1.0f);
}

__global__ void k_dinv(float* __restrict__ deg, int n) {
    int i = blockIdx.x * blockDim.x + threadIdx.x;
    if (i < n) {
        float d = deg[i];
        deg[i] = (d > 0.0f) ? rsqrtf(d) : 0.0f;
    }
}

// one wave (32 lanes) per edge, 4 features per lane (float4 gather + 4 f32 atomics)
__global__ void k_scatter(const float* __restrict__ x, const int* __restrict__ rowIdx,
                          const int* __restrict__ colIdx, const float* __restrict__ dinv,
                          float* __restrict__ ax, int E) {
    long gid  = (long)blockIdx.x * blockDim.x + threadIdx.x;
    int  e    = (int)(gid >> 5);
    int  lane = (int)(gid & 31);
    if (e >= E) return;
    int r = rowIdx[e], c = colIdx[e];
    float nrm = dinv[r] * dinv[c];
    float4 v = ((const float4*)(x + (long)r * 128))[lane];
    float* dst = ax + (long)c * 128 + lane * 4;
    atomicAdd(dst + 0, nrm * v.x);
    atomicAdd(dst + 1, nrm * v.y);
    atomicAdd(dst + 2, nrm * v.z);
    atomicAdd(dst + 3, nrm * v.w);
}

// ---------------------------------------------------------------- weight packing
// Packs W[K][Nc] (f32 row-major) into per-lane WMMA B-fragment order (bf16):
//   out[(((kt*NTP)+nt)*32 + lane)*16 + i] = B[kt*32 + (lane>>4)*16 + i][nt*16 + (lane&15)]
__global__ void k_pack_w(const float* __restrict__ W, int K, int Nc, int NTP,
                         __bf16* __restrict__ out) {
    int total = (K >> 5) * NTP * 512;
    for (int idx = blockIdx.x * blockDim.x + threadIdx.x; idx < total;
         idx += gridDim.x * blockDim.x) {
        int i    = idx & 15;
        int lane = (idx >> 4) & 31;
        int tile = idx >> 9;
        int nt   = tile % NTP;
        int kt   = tile / NTP;
        int k    = kt * 32 + (lane >> 4) * 16 + i;
        int n    = nt * 16 + (lane & 15);
        float v  = (n < Nc) ? W[(long)k * Nc + n] : 0.0f;
        out[idx] = (__bf16)v;
    }
}

// ---------------------------------------------------------------- BN stats
__global__ void k_bn_stats(const float* __restrict__ T, int M, float* __restrict__ stats) {
    // blockDim.x == 256 == H; column = threadIdx.x
    int  c   = threadIdx.x;
    long rpb = (M + gridDim.x - 1) / gridDim.x;
    long r0  = (long)blockIdx.x * rpb;
    long r1  = r0 + rpb; if (r1 > M) r1 = M;
    float s = 0.0f, s2 = 0.0f;
    for (long r = r0; r < r1; ++r) {
        float v = T[r * 256 + c];
        s += v; s2 += v * v;
    }
    atomicAdd(&stats[c],       s);
    atomicAdd(&stats[256 + c], s2);
}

__global__ void k_bn_finalize(const float* __restrict__ g, const float* __restrict__ b,
                              float* __restrict__ stats, float invN) {
    int c = threadIdx.x;   // 256 threads
    float mean = stats[c] * invN;
    float var  = stats[256 + c] * invN - mean * mean;
    float sc   = g[c] * rsqrtf(var + 1e-5f);
    stats[512 + c] = sc;
    stats[768 + c] = b[c] - mean * sc;
}

// ---------------------------------------------------------------- WMMA GEMM
// C[M x Nc] = act(A[M x K]) @ Bpacked + bias, A f32 row-major (lda),
// optional fused BN(scale,shift)+ReLU during f32->bf16 LDS staging.
// Block: 256 thr = 8 waves (4x2 wave grid); block tile 128 x (32*NI);
// wave tile 32 x (16*NI): 2 m-frags x NI n-frags of 16x16x32 bf16 WMMA.
template <int NI>
__launch_bounds__(256)
__global__ void k_wmma_gemm(const float* __restrict__ A, int lda, int M, int K,
                            const __bf16* __restrict__ Bp, int NTP,
                            const float* __restrict__ bias,
                            const float* __restrict__ bnScale,
                            const float* __restrict__ bnShift, int useBn,
                            float* __restrict__ C, long ldc, int Nc) {
    constexpr int BM = 128, AS = 40;            // LDS row stride: 40 bf16 = 80B = 5x16B
    __shared__ __bf16 As[BM * AS];

    const int tid   = threadIdx.x;
    const int wave  = tid >> 5, lane = tid & 31;
    const int wr    = wave & 3, wc = wave >> 2;     // 4x2 wave grid
    const int l15   = lane & 15, khalf = lane >> 4;
    const long blockM  = (long)blockIdx.x * BM;
    const int  blockNt = blockIdx.y * 2 * NI;       // base 16-col n-tile of block

    v8f acc[2][NI] = {};

    union UA { uint4 q[2]; bf16x16 v; };
    union UB { uint4 q[2]; bf16x16 v; };

    const int kTiles = K >> 5;
    for (int kt = 0; kt < kTiles; ++kt) {
        const int k0 = kt << 5;

        // prefetch next K-tile of A (global_prefetch_b8)
        if (kt + 1 < kTiles) {
            long prow = blockM + (tid >> 1);
            if (prow < M)
                __builtin_prefetch(A + prow * lda + k0 + 32 + (tid & 1) * 16, 0, 3);
        }

        // ---- stage A tile (128x32) f32 -> bf16 LDS, fused BN+ReLU ----
        #pragma unroll
        for (int it = 0; it < 4; ++it) {
            int  e4   = tid + it * 256;             // float4 id; 8 per row
            int  rrow = e4 >> 3;
            int  c4   = e4 & 7;
            long grow = blockM + rrow;
            float4 v = make_float4(0.f, 0.f, 0.f, 0.f);
            if (grow < M) v = *(const float4*)(A + grow * lda + k0 + c4 * 4);
            if (useBn) {
                int kb = k0 + c4 * 4;
                v.x = fmaxf(0.f, v.x * bnScale[kb + 0] + bnShift[kb + 0]);
                v.y = fmaxf(0.f, v.y * bnScale[kb + 1] + bnShift[kb + 1]);
                v.z = fmaxf(0.f, v.z * bnScale[kb + 2] + bnShift[kb + 2]);
                v.w = fmaxf(0.f, v.w * bnScale[kb + 3] + bnShift[kb + 3]);
            }
            union { __bf16 b[4]; uint2 u; } cv;
            cv.b[0] = (__bf16)v.x; cv.b[1] = (__bf16)v.y;
            cv.b[2] = (__bf16)v.z; cv.b[3] = (__bf16)v.w;
            *(uint2*)(As + rrow * AS + c4 * 4) = cv.u;
        }
        __syncthreads();

        // ---- A fragments: two ds_load_b128 each (layout is dword-contiguous) ----
        const uint4* AsQ = (const uint4*)As;        // 5 uint4 per LDS row
        UA af[2];
        #pragma unroll
        for (int mi = 0; mi < 2; ++mi) {
            int row = wr * 32 + mi * 16 + l15;
            af[mi].q[0] = AsQ[row * 5 + khalf * 2];     // K = khalf*16 + 0..7
            af[mi].q[1] = AsQ[row * 5 + khalf * 2 + 1]; // K = khalf*16 + 8..15
        }
        // ---- B fragments (pre-packed, contiguous 32B per lane) + WMMA ----
        #pragma unroll
        for (int ni = 0; ni < NI; ++ni) {
            int nt = blockNt + wc * NI + ni;
            UB bfr;
            if (nt < NTP) {
                const uint4* src =
                    (const uint4*)(Bp + (((long)kt * NTP + nt) * 32 + lane) * 16);
                bfr.q[0] = src[0];
                bfr.q[1] = src[1];
            } else {
                bfr.q[0] = make_uint4(0, 0, 0, 0);
                bfr.q[1] = make_uint4(0, 0, 0, 0);
            }
            acc[0][ni] = __builtin_amdgcn_wmma_f32_16x16x32_bf16(
                false, af[0].v, false, bfr.v, (short)0, acc[0][ni], false, false);
            acc[1][ni] = __builtin_amdgcn_wmma_f32_16x16x32_bf16(
                false, af[1].v, false, bfr.v, (short)0, acc[1][ni], false, false);
        }
        __syncthreads();
    }

    // ---- epilogue: +bias, strided store (enables free concat) ----
    #pragma unroll
    for (int ni = 0; ni < NI; ++ni) {
        int col = (blockNt + wc * NI + ni) * 16 + l15;
        if (col >= Nc) continue;
        float bv = bias[col];
        #pragma unroll
        for (int mi = 0; mi < 2; ++mi) {
            long rbase = blockM + wr * 32 + mi * 16 + khalf * 8;
            #pragma unroll
            for (int v = 0; v < 8; ++v) {
                long r = rbase + v;
                if (r < M) C[r * ldc + col] = acc[mi][ni][v] + bv;
            }
        }
    }
}

// ---------------------------------------------------------------- launcher
extern "C" void kernel_launch(void* const* d_in, const int* in_sizes, int n_in,
                              void* d_out, int out_size, void* d_ws, size_t ws_size,
                              hipStream_t stream) {
    const int N = 100000, E = 600000, D = 128, H = 256, O = 40;
    const float* x   = (const float*)d_in[0];
    const int*   ei  = (const int*)d_in[1];
    const float *Wx1 = (const float*)d_in[2],  *bx1 = (const float*)d_in[3];
    const float *gxp = (const float*)d_in[4],  *btx = (const float*)d_in[5];
    const float *Wx2 = (const float*)d_in[6],  *bx2 = (const float*)d_in[7];
    const float *Wa1 = (const float*)d_in[8],  *ba1 = (const float*)d_in[9];
    const float *gap = (const float*)d_in[10], *bta = (const float*)d_in[11];
    const float *Wa2 = (const float*)d_in[12], *ba2 = (const float*)d_in[13];
    const float *Wf1 = (const float*)d_in[14], *bf1 = (const float*)d_in[15];
    const float *gfp = (const float*)d_in[16], *btf = (const float*)d_in[17];
    const float *Wf2 = (const float*)d_in[18], *bf2 = (const float*)d_in[19];
    float* out = (float*)d_out;

    // ---- carve workspace (deterministic offsets) ----
    char*  ws  = (char*)d_ws;
    size_t off = 0;
    auto alloc = [&](size_t bytes) -> void* {
        void* p = ws + off;
        off += (bytes + 255) & ~(size_t)255;
        return p;
    };
    float*  deg   = (float*)alloc((size_t)N * 4);
    float*  ax    = (float*)alloc((size_t)N * D * 4);
    float*  t1    = (float*)alloc((size_t)N * H * 4);
    float*  cat   = (float*)alloc((size_t)N * 2 * H * 4);
    float*  stats = (float*)alloc((size_t)4 * H * 4);
    __bf16* wx1p  = (__bf16*)alloc((size_t)(D / 32) * 16 * 512 * 2);
    __bf16* wx2p  = (__bf16*)alloc((size_t)(H / 32) * 16 * 512 * 2);
    __bf16* wa1p  = (__bf16*)alloc((size_t)(D / 32) * 16 * 512 * 2);
    __bf16* wa2p  = (__bf16*)alloc((size_t)(H / 32) * 16 * 512 * 2);
    __bf16* wf1p  = (__bf16*)alloc((size_t)(2 * H / 32) * 16 * 512 * 2);
    __bf16* wf2p  = (__bf16*)alloc((size_t)(H / 32) * 3 * 512 * 2);

    dim3 blk(256);

    // graph normalization + scatter
    k_zero<<<512,  blk, 0, stream>>>(deg, (long)N);
    k_zero<<<2048, blk, 0, stream>>>(ax, (long)N * D);
    k_degree<<<(E + 255) / 256, blk, 0, stream>>>(ei + E, deg, E);
    k_dinv<<<(N + 255) / 256, blk, 0, stream>>>(deg, N);
    k_scatter<<<(int)(((long)E * 32 + 255) / 256), blk, 0, stream>>>(x, ei, ei + E, deg, ax, E);

    // pack weights into WMMA B-fragment layout (bf16)
    k_pack_w<<<64,  blk, 0, stream>>>(Wx1, D,     H, 16, wx1p);
    k_pack_w<<<128, blk, 0, stream>>>(Wx2, H,     H, 16, wx2p);
    k_pack_w<<<64,  blk, 0, stream>>>(Wa1, D,     H, 16, wa1p);
    k_pack_w<<<128, blk, 0, stream>>>(Wa2, H,     H, 16, wa2p);
    k_pack_w<<<256, blk, 0, stream>>>(Wf1, 2 * H, H, 16, wf1p);
    k_pack_w<<<48,  blk, 0, stream>>>(Wf2, H,     O, 3,  wf2p);

    const int gm = (N + 127) / 128;   // 782
    dim3 gridN4(gm, 2);               // NI=4 : 128 cols/block, grid.y = 256/128
    dim3 gridN8(gm, 1);               // NI=8 : 256 cols/block (single pass over A)
    dim3 gridN2(gm, 1);               // NI=2 : 64 cols/block  (final 40-col GEMM)
    const float invN = 1.0f / (float)N;

    // ---- mlp_x : t1 = x@Wx1+b ; BN stats ; cat[:, :H] = relu(bn(t1))@Wx2+b ----
    k_wmma_gemm<4><<<gridN4, blk, 0, stream>>>(x, D, N, D, wx1p, 16, bx1,
                                               nullptr, nullptr, 0, t1, H, H);
    k_zero<<<8, blk, 0, stream>>>(stats, 2 * H);
    k_bn_stats<<<512, blk, 0, stream>>>(t1, N, stats);
    k_bn_finalize<<<1, blk, 0, stream>>>(gxp, btx, stats, invN);
    k_wmma_gemm<4><<<gridN4, blk, 0, stream>>>(t1, H, N, H, wx2p, 16, bx2,
                                               stats + 2 * H, stats + 3 * H, 1,
                                               cat, 2 * H, H);

    // ---- mlp_a : same on ax, output into cat[:, H:2H] ----
    k_wmma_gemm<4><<<gridN4, blk, 0, stream>>>(ax, D, N, D, wa1p, 16, ba1,
                                               nullptr, nullptr, 0, t1, H, H);
    k_zero<<<8, blk, 0, stream>>>(stats, 2 * H);
    k_bn_stats<<<512, blk, 0, stream>>>(t1, N, stats);
    k_bn_finalize<<<1, blk, 0, stream>>>(gap, bta, stats, invN);
    k_wmma_gemm<4><<<gridN4, blk, 0, stream>>>(t1, H, N, H, wa2p, 16, ba2,
                                               stats + 2 * H, stats + 3 * H, 1,
                                               cat + H, 2 * H, H);

    // ---- mlp_final : t1 = cat@Wf1+b (K=512, cat > L2 -> single pass, NI=8) ----
    k_wmma_gemm<8><<<gridN8, blk, 0, stream>>>(cat, 2 * H, N, 2 * H, wf1p, 16, bf1,
                                               nullptr, nullptr, 0, t1, H, H);
    k_zero<<<8, blk, 0, stream>>>(stats, 2 * H);
    k_bn_stats<<<512, blk, 0, stream>>>(t1, N, stats);
    k_bn_finalize<<<1, blk, 0, stream>>>(gfp, btf, stats, invN);
    k_wmma_gemm<2><<<gridN2, blk, 0, stream>>>(t1, H, N, H, wf2p, 3, bf2,
                                               stats + 2 * H, stats + 3 * H, 1,
                                               out, O, O);
}
